// MultiHeadAttention_68702296867318
// MI455X (gfx1250) — matrix operand order repaired and checked by reading, not compiled, not run on hardware
//
#include <hip/hip_runtime.h>

// MI455X / gfx1250: wave32, WMMA (16x16x32 bf16 -> f32 accum)
typedef __attribute__((ext_vector_type(16))) __bf16 v16bf;
typedef __attribute__((ext_vector_type(8)))  float  v8f;

// Hardware f32->bf16 convert (RNE): let clang pick v_cvt_*bf16* (incl. packed form)
__device__ __forceinline__ unsigned short f2bf(float f) {
  __bf16 h = (__bf16)f;
  return __builtin_bit_cast(unsigned short, h);
}

enum { MODE_PROJ_QK = 0, MODE_PROJ_VT = 1, MODE_SCORES = 2, MODE_AV = 3, MODE_FINAL = 4 };

template <typename T> struct IsF32 { static constexpr bool v = false; };
template <> struct IsF32<float>    { static constexpr bool v = true;  };

union FragU { uint4 u4[2]; v16bf v; };   // 32 bytes: one WMMA A/B operand (8 VGPRs)

// Generic bf16 WMMA GEMM: C[M,N] = A[M,K] * B^T  with B stored [N][K] bf16 row-major.
//   A: f32 (converted to bf16 while staging to LDS) or bf16 depending on AT.
//   Block tile 128 x (128|64), K-step 32, double-buffered LDS, 8 waves (2x4 grid).
//   Pipeline per step: global-load regs(i+1) -> prefetch(i+2) -> wmma(i) -> lds-store(i+1) -> barrier.
template <int MODE, typename AT>
__global__ __launch_bounds__(256) void gemm_k(
    const AT* __restrict__ Aall, const unsigned short* __restrict__ Ball,
    float* __restrict__ outF, unsigned short* __restrict__ outB,
    const float* __restrict__ bias, int K, float scale)
{
  constexpr int BM  = 128;
  constexpr int BN  = (MODE == MODE_AV) ? 64 : 128;
  constexpr int WN  = BN / 4;           // per-wave N extent (2x4 wave grid)
  constexpr int MF  = 4;                // 64 / 16 M-fragments per wave
  constexpr int NF  = WN / 16;          // N-fragments per wave (2 or 1)
  constexpr int LDA = 40;               // ushorts per LDS row: 32 data + 8 pad (80B, 16B-aligned chunks)
  constexpr int BP  = BN / 32;          // B staging passes (4 or 2)
  constexpr int ABUF = BM * LDA;        // ushorts per A buffer
  constexpr int BBUF = BN * LDA;        // ushorts per B buffer

  __shared__ unsigned short As[2 * ABUF];
  __shared__ unsigned short Bs[2 * BBUF];

  const int tid  = threadIdx.x;
  const int wave = tid >> 5;
  const int lane = tid & 31;
  const int half = lane >> 4;           // half-wave select (ISA 7.12.2 layouts)
  const int mn   = lane & 15;
  const int wm   = wave >> 2;           // 0..1
  const int wn   = wave & 3;            // 0..3

  const int z = blockIdx.z;             // batch*head index for per-(b,h) GEMMs
  size_t aoff = 0, boff = 0;
  if constexpr (MODE == MODE_SCORES) { aoff = (size_t)z * 65536; boff = (size_t)z * 65536; }
  if constexpr (MODE == MODE_AV)     { int h = z & 15, b = z >> 4;
                                       aoff = ((size_t)(h * 4 + b)) << 20;   // attns row-major [h*B+b]
                                       boff = (size_t)z * 65536; }

  const AT*             Atile = Aall + aoff + (size_t)blockIdx.y * BM * K;
  const unsigned short* Btile = Ball + boff + (size_t)blockIdx.x * BN * K;

  v8f vzero = {};
  v8f acc[MF][NF];
  #pragma unroll
  for (int f = 0; f < MF; ++f)
    #pragma unroll
    for (int g = 0; g < NF; ++g) acc[f][g] = vzero;

  // ---- register staging for the in-flight tile ----
  float4  aF[4];
  ushort4 aS[4];
  ushort4 bS[BP];

  auto gload = [&](int step) {                 // issue global loads for tile `step`
    const int k0 = step << 5;
    #pragma unroll
    for (int p = 0; p < 4; ++p) {
      int li = p * 256 + tid; int row = li >> 3; int kc = (li & 7) << 2;
      if constexpr (IsF32<AT>::v)
        aF[p] = *reinterpret_cast<const float4*>(Atile + (size_t)row * K + k0 + kc);
      else
        aS[p] = *reinterpret_cast<const ushort4*>(Atile + (size_t)row * K + k0 + kc);
    }
    #pragma unroll
    for (int p = 0; p < BP; ++p) {
      int li = p * 256 + tid; int row = li >> 3; int kc = (li & 7) << 2;
      bS[p] = *reinterpret_cast<const ushort4*>(Btile + (size_t)row * K + k0 + kc);
    }
  };

  auto pref = [&](int step) {                  // global_prefetch_b8 for tile `step`
    const int k0 = step << 5;
    #pragma unroll
    for (int p = 0; p < 4; ++p) {
      int li = p * 256 + tid; int row = li >> 3; int kc = (li & 7) << 2;
      __builtin_prefetch(Atile + (size_t)row * K + k0 + kc, 0, 3);
    }
    #pragma unroll
    for (int p = 0; p < BP; ++p) {
      int li = p * 256 + tid; int row = li >> 3; int kc = (li & 7) << 2;
      __builtin_prefetch(Btile + (size_t)row * K + k0 + kc, 0, 3);
    }
  };

  auto lstore = [&](int buf) {                 // staged regs -> LDS buffer `buf`
    const int ab = buf * ABUF, bb = buf * BBUF;
    #pragma unroll
    for (int p = 0; p < 4; ++p) {
      int li = p * 256 + tid; int row = li >> 3; int kc = (li & 7) << 2;
      ushort4 sv;
      if constexpr (IsF32<AT>::v) {
        sv.x = f2bf(aF[p].x); sv.y = f2bf(aF[p].y); sv.z = f2bf(aF[p].z); sv.w = f2bf(aF[p].w);
      } else sv = aS[p];
      *reinterpret_cast<ushort4*>(&As[ab + row * LDA + kc]) = sv;
    }
    #pragma unroll
    for (int p = 0; p < BP; ++p) {
      int li = p * 256 + tid; int row = li >> 3; int kc = (li & 7) << 2;
      *reinterpret_cast<ushort4*>(&Bs[bb + row * LDA + kc]) = bS[p];
    }
  };

  auto compute = [&](int buf) {                // 16x16x32 WMMAs from LDS buffer `buf`
    const int ab = buf * ABUF, bb = buf * BBUF;
    // B fragments: lane holds N=mn, K = half*16 .. half*16+15 (contiguous 32B)
    FragU bf[NF];
    #pragma unroll
    for (int g = 0; g < NF; ++g) {
      const unsigned short* src = &Bs[bb + (wn * WN + g * 16 + mn) * LDA + half * 16];
      bf[g].u4[0] = *reinterpret_cast<const uint4*>(src);
      bf[g].u4[1] = *reinterpret_cast<const uint4*>(src + 8);
    }
    // A fragments: lane holds M=mn, K = half*16 .. half*16+15 (contiguous 32B)
    #pragma unroll
    for (int f = 0; f < MF; ++f) {
      FragU af;
      const unsigned short* src = &As[ab + (wm * 64 + f * 16 + mn) * LDA + half * 16];
      af.u4[0] = *reinterpret_cast<const uint4*>(src);
      af.u4[1] = *reinterpret_cast<const uint4*>(src + 8);
      #pragma unroll
      for (int g = 0; g < NF; ++g)
        acc[f][g] = __builtin_amdgcn_wmma_f32_16x16x32_bf16(
            false, af.v, false, bf[g].v, (short)0, acc[f][g], false, false);
    }
  };

  // ---- double-buffered main loop ----
  const int nsteps = K >> 5;
  gload(0);
  lstore(0);
  __syncthreads();
  for (int i = 0; i < nsteps; ++i) {
    if (i + 1 < nsteps) gload(i + 1);          // loads in flight during compute
    if (i + 2 < nsteps) pref(i + 2);           // L2/WGP prefetch two tiles ahead
    compute(i & 1);
    if (i + 1 < nsteps) lstore((i + 1) & 1);   // s_wait_loadcnt lands here, after WMMAs
    __syncthreads();
  }

  // ---- store: C layout (ISA): m = half*8 + r, n = mn within each 16x16 frag ----
  const int tM = blockIdx.y * BM + wm * 64;
  const int tN = blockIdx.x * BN + wn * WN;
  #pragma unroll
  for (int f = 0; f < MF; ++f) {
    #pragma unroll
    for (int g = 0; g < NF; ++g) {
      #pragma unroll
      for (int r = 0; r < 8; ++r) {
        int m = tM + f * 16 + half * 8 + r;
        int n = tN + g * 16 + mn;
        float val = acc[f][g][r];
        if constexpr (MODE == MODE_PROJ_QK) {
          // q/k buffers: [b][h][s][64] bf16
          size_t idx = ((size_t)((m >> 10) * 16 + (n >> 6)) << 16)
                     + (size_t)((m & 1023) << 6) + (n & 63);
          outB[idx] = f2bf(val);
        } else if constexpr (MODE == MODE_PROJ_VT) {
          // v buffer transposed: [b][h][v][s] bf16 (so attn@V sees B as [N][K])
          size_t idx = ((size_t)((m >> 10) * 16 + (n >> 6)) << 16)
                     + (size_t)((n & 63) << 10) + (m & 1023);
          outB[idx] = f2bf(val);
        } else if constexpr (MODE == MODE_SCORES) {
          int h = z & 15, b = z >> 4;
          outF[(((size_t)(h * 4 + b)) << 20) + (size_t)m * 1024 + n] = val * scale;
        } else if constexpr (MODE == MODE_AV) {
          int h = z & 15, b = z >> 4;
          outB[((size_t)b << 20) + (size_t)m * 1024 + h * 64 + n] = f2bf(val);
        } else { // MODE_FINAL
          outF[(size_t)m * 1024 + n] = val + bias[n];
        }
      }
    }
  }
}

// Row softmax in place over attns rows of length 1024 (one block per row).
__global__ __launch_bounds__(256) void softmax_k(float* __restrict__ attns) {
  __shared__ float red[256];
  float* p = attns + ((size_t)blockIdx.x << 10);
  const int tid = threadIdx.x;
  float4 x = reinterpret_cast<float4*>(p)[tid];
  float m = fmaxf(fmaxf(x.x, x.y), fmaxf(x.z, x.w));
  red[tid] = m; __syncthreads();
  for (int s = 128; s > 0; s >>= 1) { if (tid < s) red[tid] = fmaxf(red[tid], red[tid + s]); __syncthreads(); }
  m = red[0]; __syncthreads();
  float4 e;
  e.x = __expf(x.x - m); e.y = __expf(x.y - m);
  e.z = __expf(x.z - m); e.w = __expf(x.w - m);
  red[tid] = e.x + e.y + e.z + e.w; __syncthreads();
  for (int s = 128; s > 0; s >>= 1) { if (tid < s) red[tid] += red[tid + s]; __syncthreads(); }
  float inv = 1.0f / red[0];
  e.x *= inv; e.y *= inv; e.z *= inv; e.w *= inv;
  reinterpret_cast<float4*>(p)[tid] = e;
}

// w_{q,k,v}[H,D,64] -> Wt[N=H*64][K=D] bf16 (B operand layout for the projection GEMMs)
__global__ __launch_bounds__(256) void pack_wqkv_k(const float* __restrict__ w,
                                                   unsigned short* __restrict__ wt) {
  int i = blockIdx.x * 256 + threadIdx.x;   // i = n*1024 + d, over 1M elems
  int n = i >> 10, d = i & 1023;
  wt[i] = f2bf(w[((size_t)(n >> 6) << 16) + ((size_t)d << 6) + (n & 63)]);
}

// w_proj[D, H*DV] is already [N][K] for out = X @ w_proj^T: just convert to bf16.
__global__ __launch_bounds__(256) void pack_f32_bf16_k(const float* __restrict__ w,
                                                       unsigned short* __restrict__ wt) {
  int i = blockIdx.x * 256 + threadIdx.x;
  wt[i] = f2bf(w[i]);
}

extern "C" void kernel_launch(void* const* d_in, const int* in_sizes, int n_in,
                              void* d_out, int out_size, void* d_ws, size_t ws_size,
                              hipStream_t stream) {
  const float* query = (const float*)d_in[0];
  const float* key   = (const float*)d_in[1];
  const float* value = (const float*)d_in[2];
  const float* w_q   = (const float*)d_in[3];
  const float* w_k   = (const float*)d_in[4];
  const float* w_v   = (const float*)d_in[5];
  const float* w_pj  = (const float*)d_in[6];
  const float* b_pj  = (const float*)d_in[7];

  float* out   = (float*)d_out;                      // [4,1024,1024] f32
  float* attns = out + (size_t)4 * 1024 * 1024;      // [64,1024,1024] f32

  // workspace layout (bf16 elements), 40 MB total
  unsigned short* ws   = (unsigned short*)d_ws;
  const size_t M1 = 1024 * 1024;
  unsigned short* wq_t = ws;                 // 1M
  unsigned short* wk_t = ws + 1 * M1;        // 1M
  unsigned short* wv_t = ws + 2 * M1;        // 1M
  unsigned short* wp_t = ws + 3 * M1;        // 1M
  unsigned short* qbuf = ws + 4 * M1;        // 4M  [b][h][s][64]
  unsigned short* kbuf = ws + 8 * M1;        // 4M  [b][h][s][64]
  unsigned short* vtb  = ws + 12 * M1;       // 4M  [b][h][v][s]
  unsigned short* hcat = ws + 16 * M1;       // 4M  [b][s][h*64+v]

  // 0) pack weights into [N][K] bf16
  pack_wqkv_k<<<4096, 256, 0, stream>>>(w_q, wq_t);
  pack_wqkv_k<<<4096, 256, 0, stream>>>(w_k, wk_t);
  pack_wqkv_k<<<4096, 256, 0, stream>>>(w_v, wv_t);
  pack_f32_bf16_k<<<4096, 256, 0, stream>>>(w_pj, wp_t);

  // 1) projections: [4096,1024] x [1024,1024] (all heads fused)
  gemm_k<MODE_PROJ_QK, float><<<dim3(8, 32, 1), 256, 0, stream>>>(query, wq_t, nullptr, qbuf, nullptr, 1024, 1.f);
  gemm_k<MODE_PROJ_QK, float><<<dim3(8, 32, 1), 256, 0, stream>>>(key,   wk_t, nullptr, kbuf, nullptr, 1024, 1.f);
  gemm_k<MODE_PROJ_VT, float><<<dim3(8, 32, 1), 256, 0, stream>>>(value, wv_t, nullptr, vtb,  nullptr, 1024, 1.f);

  // 2) scores = Q K^T / 32 per (b,h): [1024,64] x [1024,64]^T -> f32 attns
  gemm_k<MODE_SCORES, unsigned short><<<dim3(8, 8, 64), 256, 0, stream>>>(qbuf, kbuf, attns, nullptr, nullptr, 64, 0.03125f);

  // 3) softmax rows in place
  softmax_k<<<65536, 256, 0, stream>>>(attns);

  // 4) out_head = attn @ V per (b,h): [1024,1024] x [64,1024]^T -> bf16 head-concat
  gemm_k<MODE_AV, float><<<dim3(1, 8, 64), 256, 0, stream>>>(attns, vtb, nullptr, hcat, nullptr, 1024, 1.f);

  // 5) final: [4096,1024] x w_proj^T + bias -> d_out
  gemm_k<MODE_FINAL, unsigned short><<<dim3(8, 32, 1), 256, 0, stream>>>(hcat, wp_t, out, nullptr, b_pj, 1024, 1.f);
}
